// GCNNet_80633716015157
// MI455X (gfx1250) — matrix operand order repaired and checked by reading, not compiled, not run on hardware
//
#include <hip/hip_runtime.h>
#include <hip/hip_bf16.h>
#include <stdint.h>

// ---------- WMMA types ----------
typedef __attribute__((ext_vector_type(16))) __bf16 v16bf;
typedef __attribute__((ext_vector_type(8)))  float  v8f;
typedef __attribute__((ext_vector_type(8)))  unsigned short us8;

typedef __attribute__((ext_vector_type(4))) unsigned int u32x4;
typedef __attribute__((ext_vector_type(8))) int i32x8;
typedef __attribute__((ext_vector_type(4))) int i32x4;

union BF16x16 { v16bf v; us8 h[2]; unsigned short u[16]; };

__device__ inline unsigned short f2bf(float f) {
  unsigned int u = __float_as_uint(f);
  unsigned int r = u + 0x7FFFu + ((u >> 16) & 1u);   // round-to-nearest-even
  return (unsigned short)(r >> 16);
}

// ---------- Tensor Data Mover: 2D bf16 tile -> LDS ----------
// D# per CDNA5 ISA 8.3/8.4. tile_d0 elems/row, tile_d1 rows, tensor row stride
// stride0 (elements). Issue from ONE wave only; wave-level op (EXEC ignored).
// This toolchain exposes the 6-arg builtin form (extra i32x8 group before cpol).
__device__ inline void tdm_load_tile_bf16(unsigned lds_byte_off, const void* gptr,
                                          unsigned tile_d0, unsigned tile_d1,
                                          unsigned tensor_d0, unsigned tensor_d1,
                                          unsigned long long stride0) {
  unsigned long long ga = (unsigned long long)gptr;
  u32x4 g0;
  g0[0] = 1u;                                                  // count=1, user mode
  g0[1] = lds_byte_off;                                        // lds_addr
  g0[2] = (unsigned)ga;                                        // global_addr[31:0]
  g0[3] = (unsigned)((ga >> 32) & 0x01FFFFFFu) | (2u << 30);   // addr[56:32] | type=2
  i32x8 g1;
  g1[0] = (int)(1u << 16);                                     // data_size=1 (2B), no mcast
  g1[1] = (int)((tensor_d0 & 0xFFFFu) << 16);                  // tensor_dim0[15:0]
  g1[2] = (int)((tensor_d0 >> 16) | ((tensor_d1 & 0xFFFFu) << 16));
  g1[3] = (int)((tensor_d1 >> 16) | ((tile_d0 & 0xFFFFu) << 16));
  g1[4] = (int)(tile_d1 & 0xFFFFu);                            // tile_dim1, tile_dim2=0
  g1[5] = (int)(unsigned)(stride0 & 0xFFFFFFFFull);            // dim0_stride[31:0]
  g1[6] = (int)(unsigned)((stride0 >> 32) & 0xFFFFull);        // dim0_stride[47:32]
  g1[7] = 0;
  i32x4 z4 = {0, 0, 0, 0};
  i32x8 z8 = {0, 0, 0, 0, 0, 0, 0, 0};
  __builtin_amdgcn_tensor_load_to_lds(g0, g1, z4, z4, z8, 0);
}

__device__ inline unsigned lds_offset_of(const void* p) {
  return (unsigned)(unsigned long long)p;   // generic LDS addr: low 32 bits = LDS offset
}

// A fragment: 16x32 bf16 row-major LDS tile; two contiguous 16B halves per lane.
__device__ inline void load_frag_a(BF16x16& a, const unsigned short* sA, int lda,
                                   int row0, int kc, int lane) {
  int m  = row0 + (lane & 15);
  int kb = (lane < 16) ? 0 : 8;
  const unsigned short* p = sA + m * lda + kc + kb;
  a.h[0] = *(const us8*)(p);
  a.h[1] = *(const us8*)(p + 16);
}

// B fragment from N-major (transposed) LDS tile sBt[n][k]; two 16B halves per lane (k = kb, kb+16).
__device__ inline void load_frag_bt(BF16x16& b, const unsigned short* sBt, int ldk,
                                    int kc, int col0, int lane) {
  int n  = col0 + (lane & 15);
  int kb = (lane < 16) ? 0 : 8;
  const unsigned short* p = sBt + n * ldk + kc + kb;
  b.h[0] = *(const us8*)(p);
  b.h[1] = *(const us8*)(p + 16);
}

// ---------- small helpers ----------
// convert f32 (K x N) -> bf16 transposed (N x K)
__global__ void cvt_bf16_t_kernel(const float* __restrict__ in, unsigned short* __restrict__ out,
                                  int K, int N) {
  int i = blockIdx.x * 256 + threadIdx.x;
  if (i < K * N) {
    int k = i / N, n = i - k * N;
    out[n * K + k] = f2bf(in[i]);
  }
}

__global__ void degree_kernel(const int* __restrict__ src, const int* __restrict__ dst,
                              float* __restrict__ outdeg, float* __restrict__ indeg, int E) {
  int e = blockIdx.x * 256 + threadIdx.x;
  if (e < E) {
    atomicAdd(&outdeg[src[e]], 1.0f);
    atomicAdd(&indeg[dst[e]], 1.0f);
  }
}

__global__ void norm_kernel(const float* __restrict__ outdeg, const float* __restrict__ indeg,
                            float* __restrict__ ns, float* __restrict__ nd, int Nn) {
  int i = blockIdx.x * 256 + threadIdx.x;
  if (i < Nn) {
    float od = outdeg[i], id = indeg[i];
    ns[i] = od > 0.f ? rsqrtf(od) : 0.f;
    nd[i] = id > 0.f ? rsqrtf(id) : 0.f;
  }
}

// agg[dst] += x[src] * norm_src[src]  (L2-resident scatter-add)
__global__ void scatter_kernel(const float* __restrict__ x, const int* __restrict__ src,
                               const int* __restrict__ dst, const float* __restrict__ ns,
                               float* __restrict__ agg, int E) {
  int tid = blockIdx.x * 256 + threadIdx.x;
  int e = tid >> 7;
  int f = tid & 127;
  if (e < E) {
    int s = src[e], d = dst[e];
    float v = x[(long long)s * 128 + f] * ns[s];
    atomicAdd(&agg[(long long)d * 128 + f], v);
  }
}

// ---------- generic bf16 WMMA GEMM: C = (rowScale? A*rowScale : A) @ B + bias ----------
// Bt is bf16 weights TRANSPOSED (N x K). 256 threads, tile 64(M) x 64(N), BK=32.
// B tiles staged by the Tensor Data Mover; A tiles converted f32->bf16 via VGPRs.
__global__ __launch_bounds__(256) void gemm_bf16_kernel(
    const float* __restrict__ A, const unsigned short* __restrict__ Bt,
    const float* __restrict__ bias, const float* __restrict__ rowScale,
    float* __restrict__ C, int M, int N, int K) {
  __shared__ __align__(16) unsigned short sA[64 * 32];   // 64 rows x 32 K
  __shared__ __align__(16) unsigned short sBt[64 * 32];  // 64 cols(N) x 32 K

  int tid  = threadIdx.x;
  int lane = tid & 31;
  int wave = tid >> 5;
  int wm   = wave & 3;        // 4 row stripes of 16
  int wn   = wave >> 2;       // 2 col groups of 32
  int bm   = blockIdx.x * 64;
  int bn   = blockIdx.y * 64;

  v8f acc0 = {}; v8f acc1 = {};

  int ar = tid >> 2;            // 0..63 row
  int ac = (tid & 3) * 8;       // 0,8,16,24
  int gr = bm + ar;
  unsigned sBt_off = lds_offset_of(sBt);

  for (int kc = 0; kc < K; kc += 32) {
    // TDM: stage Bt 64x32 tile (row stride K) into LDS, wave 0 only
    if (tid < 32) {
      tdm_load_tile_bf16(sBt_off, Bt + (long long)bn * K + kc,
                         32u, 64u, (unsigned)K, 64u, (unsigned long long)K);
    }
    // stage A 64x32: 2x float4 global loads, convert, one us8 LDS store
    {
      float v[8] = {0.f, 0.f, 0.f, 0.f, 0.f, 0.f, 0.f, 0.f};
      if (gr < M) {
        const float4* p = (const float4*)(A + (long long)gr * K + kc + ac);
        float4 v0 = p[0], v1 = p[1];
        float sc = rowScale ? rowScale[gr] : 1.0f;
        v[0] = v0.x * sc; v[1] = v0.y * sc; v[2] = v0.z * sc; v[3] = v0.w * sc;
        v[4] = v1.x * sc; v[5] = v1.y * sc; v[6] = v1.z * sc; v[7] = v1.w * sc;
      }
      us8 pk;
#pragma unroll
      for (int j = 0; j < 8; ++j) pk[j] = f2bf(v[j]);
      *(us8*)(sA + ar * 32 + ac) = pk;
    }
    if (tid < 32) __builtin_amdgcn_s_wait_tensorcnt(0);
    __syncthreads();

    BF16x16 af, bf0, bf1;
    load_frag_a(af, sA, 32, wm * 16, 0, lane);
    load_frag_bt(bf0, sBt, 32, 0, wn * 32,      lane);
    load_frag_bt(bf1, sBt, 32, 0, wn * 32 + 16, lane);
    acc0 = __builtin_amdgcn_wmma_f32_16x16x32_bf16(false, af.v, false, bf0.v, (short)0, acc0, false, false);
    acc1 = __builtin_amdgcn_wmma_f32_16x16x32_bf16(false, af.v, false, bf1.v, (short)0, acc1, false, false);
    __syncthreads();
  }

  int mrow = bm + wm * 16 + ((lane < 16) ? 0 : 8);
  int col0 = bn + wn * 32 + (lane & 15);
  float bia0 = bias[col0], bia1 = bias[col0 + 16];
#pragma unroll
  for (int i = 0; i < 8; ++i) {
    int r = mrow + i;
    if (r < M) {
      C[(long long)r * N + col0]      = acc0[i] + bia0;
      C[(long long)r * N + col0 + 16] = acc1[i] + bia1;
    }
  }
}

// ---------- BN statistics: per-channel sum & sumsq ----------
__global__ void bn_stats_kernel(const float* __restrict__ y, float* __restrict__ stats, int Nn) {
  int c  = threadIdx.x;             // 128 threads
  int r0 = blockIdx.x * 256;
  float s = 0.f, sq = 0.f;
  for (int i = 0; i < 256; ++i) {
    int r = r0 + i;
    if (r < Nn) {
      float v = y[(long long)r * 128 + c];
      s += v; sq += v * v;
    }
  }
  atomicAdd(&stats[c], s);
  atomicAdd(&stats[128 + c], sq);
}

// x += relu(bn(y))
__global__ void bn_apply_kernel(float* __restrict__ x, const float* __restrict__ y,
                                const float* __restrict__ stats,
                                const float* __restrict__ g, const float* __restrict__ b, int Nn) {
  long long i = (long long)blockIdx.x * 256 + threadIdx.x;
  if (i < (long long)Nn * 128) {
    int c = (int)(i & 127);
    float invN = 1.0f / (float)Nn;
    float mu   = stats[c] * invN;
    float var  = stats[128 + c] * invN - mu * mu;
    float v    = (y[i] - mu) * rsqrtf(var + 1e-5f) * g[c] + b[c];
    x[i] += fmaxf(v, 0.f);
  }
}

// ---------- fused edge MLP ----------
// out = relu(relu([x[src],x[dst]] @ W0 + b0) @ W1 + b1) @ W2 + b2
// W0t: 128 x 256 (NxK bf16), W1t: 64 x 128 (NxK bf16). 64 edges / block, 256 threads.
// Weight tiles staged by TDM; activation gather via VGPRs.
__global__ __launch_bounds__(256) void edge_mlp_kernel(
    const float* __restrict__ x, const int* __restrict__ src, const int* __restrict__ dst,
    const unsigned short* __restrict__ W0t, const float* __restrict__ b0,
    const unsigned short* __restrict__ W1t, const float* __restrict__ b1,
    const float* __restrict__ W2, const float* __restrict__ b2,
    float* __restrict__ out, int E) {
  __shared__ __align__(16) unsigned short sA[64 * 256];    // 32KB; reused: z1 (16KB) + z2 (f32 16KB)
  __shared__ __align__(16) unsigned short sW0t[128 * 32];  // 8KB chunk (N x Kchunk)
  __shared__ __align__(16) unsigned short sW1t[64 * 128];  // 16KB (N x K)
  __shared__ float sW2[130];                               // W2 (64x2) + b2

  int tid  = threadIdx.x;
  int lane = tid & 31;
  int wave = tid >> 5;
  int e0   = blockIdx.x * 64;

  // TDM: whole W1t (1D, 8192 bf16 = 16KB) into LDS, wave 0
  if (tid < 32) {
    tdm_load_tile_bf16(lds_offset_of(sW1t), W1t, 8192u, 1u, 8192u, 1u, 8192ull);
  }

  // gather A0 = [x[src], x[dst]] bf16, 64x256. 8-elem aligned chunks, never straddle halves.
#pragma unroll
  for (int j = 0; j < 8; ++j) {
    int chunk = tid + j * 256;          // 0..2047
    int r  = chunk >> 5;                // 32 chunks per row
    int cc = (chunk & 31) * 8;          // 0..248
    int e = e0 + r;
    float v[8] = {0.f, 0.f, 0.f, 0.f, 0.f, 0.f, 0.f, 0.f};
    if (e < E) {
      int node = (cc < 128) ? src[e] : dst[e];
      const float4* p = (const float4*)(x + (long long)node * 128 + (cc & 127));
      float4 v0 = p[0], v1 = p[1];
      v[0] = v0.x; v[1] = v0.y; v[2] = v0.z; v[3] = v0.w;
      v[4] = v1.x; v[5] = v1.y; v[6] = v1.z; v[7] = v1.w;
    }
    us8 pk;
#pragma unroll
    for (int q = 0; q < 8; ++q) pk[q] = f2bf(v[q]);
    *(us8*)(sA + r * 256 + cc) = pk;
  }
  if (tid < 128) sW2[tid] = W2[tid];
  if (tid < 2)   sW2[128 + tid] = b2[tid];

  // ---- GEMM1: 64x256 @ (256x128) -> z1 (relu) ----
  int wm  = wave & 3;          // row stripe
  int wcb = (wave >> 2) * 64;  // col base (4 tiles of 16)
  unsigned sW0t_off = lds_offset_of(sW0t);
  v8f acc[4] = {{}, {}, {}, {}};
  for (int kc = 0; kc < 256; kc += 32) {
    // TDM: stage W0t chunk (128 N x 32 K tile from 128x256, row stride 256), wave 0
    if (tid < 32) {
      tdm_load_tile_bf16(sW0t_off, W0t + kc, 32u, 128u, 256u, 128u, 256ull);
      __builtin_amdgcn_s_wait_tensorcnt(0);   // also covers the earlier W1t load (in-order)
    }
    __syncthreads();
    BF16x16 af, bfr;
    load_frag_a(af, sA, 256, wm * 16, kc, lane);
#pragma unroll
    for (int t = 0; t < 4; ++t) {
      load_frag_bt(bfr, sW0t, 32, 0, wcb + t * 16, lane);
      acc[t] = __builtin_amdgcn_wmma_f32_16x16x32_bf16(false, af.v, false, bfr.v, (short)0, acc[t], false, false);
    }
    __syncthreads();
  }
  // z1 = relu(acc + b0) -> bf16, reuse sA (first 16KB)
  unsigned short* sZ1 = sA;          // 64 x 128 row-major
  {
    int mrow = wm * 16 + ((lane < 16) ? 0 : 8);
#pragma unroll
    for (int t = 0; t < 4; ++t) {
      int col = wcb + t * 16 + (lane & 15);
      float bia = b0[col];
#pragma unroll
      for (int i = 0; i < 8; ++i) {
        float v = fmaxf(acc[t][i] + bia, 0.f);
        sZ1[(mrow + i) * 128 + col] = f2bf(v);
      }
    }
  }
  __syncthreads();

  // ---- GEMM2: 64x128 @ (128x64) -> z2 (relu) ----
  int wn2 = (wave >> 2) * 32;        // 2 tiles of 16
  v8f acc2[2] = {{}, {}};
  for (int kc = 0; kc < 128; kc += 32) {
    BF16x16 af, bfr;
    load_frag_a(af, sZ1, 128, wm * 16, kc, lane);
#pragma unroll
    for (int t = 0; t < 2; ++t) {
      load_frag_bt(bfr, sW1t, 128, kc, wn2 + t * 16, lane);
      acc2[t] = __builtin_amdgcn_wmma_f32_16x16x32_bf16(false, af.v, false, bfr.v, (short)0, acc2[t], false, false);
    }
  }
  float* sZ2 = (float*)&sA[64 * 128]; // second 16KB of sA, 64 x 64 f32
  {
    int mrow = wm * 16 + ((lane < 16) ? 0 : 8);
#pragma unroll
    for (int t = 0; t < 2; ++t) {
      int col = wn2 + t * 16 + (lane & 15);
      float bia = b1[col];
#pragma unroll
      for (int i = 0; i < 8; ++i) {
        sZ2[(mrow + i) * 64 + col] = fmaxf(acc2[t][i] + bia, 0.f);
      }
    }
  }
  __syncthreads();

  // ---- GEMM3: 64x64 @ 64x2 -> out ----
  if (tid < 128) {
    int r = tid >> 1, c = tid & 1;
    int e = e0 + r;
    if (e < E) {
      float s = sW2[128 + c];
      const float4* zr = (const float4*)(sZ2 + r * 64);
#pragma unroll
      for (int k4 = 0; k4 < 16; ++k4) {
        float4 z = zr[k4];
        int kb = k4 * 4;
        s += z.x * sW2[(kb + 0) * 2 + c] + z.y * sW2[(kb + 1) * 2 + c]
           + z.z * sW2[(kb + 2) * 2 + c] + z.w * sW2[(kb + 3) * 2 + c];
      }
      out[(long long)e * 2 + c] = s;
    }
  }
}

// ---------- host ----------
extern "C" void kernel_launch(void* const* d_in, const int* in_sizes, int n_in,
                              void* d_out, int out_size, void* d_ws, size_t ws_size,
                              hipStream_t stream) {
  const float* h      = (const float*)d_in[0];
  const int*   src    = (const int*)d_in[1];
  const int*   dst    = (const int*)d_in[2];
  const float* emb_W  = (const float*)d_in[3];
  const float* emb_b  = (const float*)d_in[4];
  const float* conv_W = (const float*)d_in[5];
  const float* conv_b = (const float*)d_in[6];
  const float* bn_g   = (const float*)d_in[7];
  const float* bn_b   = (const float*)d_in[8];
  const float* W0     = (const float*)d_in[9];
  const float* b0     = (const float*)d_in[10];
  const float* W1     = (const float*)d_in[11];
  const float* b1     = (const float*)d_in[12];
  const float* W2     = (const float*)d_in[13];
  const float* b2     = (const float*)d_in[14];
  float* out = (float*)d_out;

  int Nn = in_sizes[0] / 128;
  int E  = in_sizes[1];
  const int L = 4;

  uint8_t* ws = (uint8_t*)d_ws;
  size_t off = 0;
  auto carve = [&](size_t bytes) -> void* {
    void* p = ws + off;
    off = (off + bytes + 255) & ~(size_t)255;
    return p;
  };
  float* x      = (float*)carve((size_t)Nn * 128 * 4);
  float* agg    = (float*)carve((size_t)Nn * 128 * 4);
  float* ybuf   = (float*)carve((size_t)Nn * 128 * 4);
  float* outdeg = (float*)carve((size_t)Nn * 4);
  float* indeg  = (float*)carve((size_t)Nn * 4);
  float* nsr    = (float*)carve((size_t)Nn * 4);
  float* ndt    = (float*)carve((size_t)Nn * 4);
  float* stats  = (float*)carve(256 * 4);
  unsigned short* embWt  = (unsigned short*)carve(128 * 128 * 2);          // N x K
  unsigned short* convWt = (unsigned short*)carve((size_t)L * 128 * 128 * 2);
  unsigned short* W0t    = (unsigned short*)carve(128 * 256 * 2);          // 128 N x 256 K
  unsigned short* W1t    = (unsigned short*)carve(64 * 128 * 2);           // 64 N x 128 K

  // degrees + norms
  (void)hipMemsetAsync(outdeg, 0, (size_t)Nn * 4, stream);
  (void)hipMemsetAsync(indeg,  0, (size_t)Nn * 4, stream);
  degree_kernel<<<(E + 255) / 256, 256, 0, stream>>>(src, dst, outdeg, indeg, E);
  norm_kernel<<<(Nn + 255) / 256, 256, 0, stream>>>(outdeg, indeg, nsr, ndt, Nn);

  // bf16 transposed weight conversion
  cvt_bf16_t_kernel<<<(128 * 128 + 255) / 256, 256, 0, stream>>>(emb_W, embWt, 128, 128);
  for (int l = 0; l < L; ++l)
    cvt_bf16_t_kernel<<<(128 * 128 + 255) / 256, 256, 0, stream>>>(
        conv_W + (size_t)l * 128 * 128, convWt + (size_t)l * 128 * 128, 128, 128);
  cvt_bf16_t_kernel<<<(256 * 128 + 255) / 256, 256, 0, stream>>>(W0, W0t, 256, 128);
  cvt_bf16_t_kernel<<<(128 * 64 + 255) / 256, 256, 0, stream>>>(W1, W1t, 128, 64);

  // x = h @ emb_W + emb_b
  {
    dim3 grid((Nn + 63) / 64, 128 / 64);
    gemm_bf16_kernel<<<grid, 256, 0, stream>>>(h, embWt, emb_b, nullptr, x, Nn, 128, 128);
  }

  // 4 GCN layers
  for (int l = 0; l < L; ++l) {
    (void)hipMemsetAsync(agg, 0, (size_t)Nn * 128 * 4, stream);
    (void)hipMemsetAsync(stats, 0, 256 * 4, stream);
    scatter_kernel<<<(E * 128 + 255) / 256, 256, 0, stream>>>(x, src, dst, nsr, agg, E);
    dim3 grid((Nn + 63) / 64, 128 / 64);
    gemm_bf16_kernel<<<grid, 256, 0, stream>>>(agg, convWt + (size_t)l * 128 * 128,
                                               conv_b + l * 128, ndt, ybuf, Nn, 128, 128);
    bn_stats_kernel<<<(Nn + 255) / 256, 128, 0, stream>>>(ybuf, stats, Nn);
    bn_apply_kernel<<<(int)(((long long)Nn * 128 + 255) / 256), 256, 0, stream>>>(
        x, ybuf, stats, bn_g + l * 128, bn_b + l * 128, Nn);
  }

  // fused edge MLP
  edge_mlp_kernel<<<(E + 63) / 64, 256, 0, stream>>>(x, src, dst, W0t, b0, W1t, b1, W2, b2, out, E);
}
